// TemMulHeadAtte_13537736917594
// MI455X (gfx1250) — matrix operand order repaired and checked by reading, not compiled, not run on hardware
//
#include <hip/hip_runtime.h>
#include <math.h>

// ---------------------------------------------------------------------------
// Temporal multi-head attention for MI455X (gfx1250, wave32, WMMA f16->f32).
// B=16 N=128 T=64 dmodel=128 H=8 dk=32 dv=16.  One workgroup per (b,n).
// ---------------------------------------------------------------------------

typedef _Float16 v4h  __attribute__((ext_vector_type(4)));
typedef _Float16 v8h  __attribute__((ext_vector_type(8)));
typedef _Float16 v16h __attribute__((ext_vector_type(16)));
typedef float    v8f  __attribute__((ext_vector_type(8)));

#define V8F_ZERO ((v8f){0.f,0.f,0.f,0.f,0.f,0.f,0.f,0.f})

static __device__ inline v16h cat16(v8h a, v8h b) {
  return __builtin_shufflevector(a, b, 0,1,2,3,4,5,6,7,8,9,10,11,12,13,14,15);
}

static __device__ inline v8f wmma_f16(v16h a, v16h b, v8f c) {
  // v_wmma_f32_16x16x32_f16
  return __builtin_amdgcn_wmma_f32_16x16x32_f16(false, a, false, b, (short)0, c, false, false);
}

// A fragment (16x32 f16, M x K), source row-major [M][K] with K contiguous.
// ISA layout: lanes 0-15 (M=lane): halves0..7 = K0..7, halves8..15 = K16..23;
//             lanes 16-31 (M=lane-16): K8..15 and K24..31.
static __device__ inline v16h load_afrag(const _Float16* s, int ld,
                                         int row0, int col0, int lh, int lm) {
  const _Float16* base = s + (size_t)(row0 + lm) * ld + col0;
  v8h lo = *(const v8h*)(base + lh * 8);
  v8h hi = *(const v8h*)(base + 16 + lh * 8);
  return cat16(lo, hi);
}

// B fragment (32x16 f16, K x N), source laid out as B^T row-major [N][K]
// (K contiguous per row).  Lanes 0-15: col N=lane, K0..7,K16..23; lanes 16-31: K8..15,K24..31.
static __device__ inline v16h load_bfrag(const _Float16* s, int ld,
                                         int ncol0, int k0, int lh, int lm) {
  const _Float16* base = s + (size_t)(ncol0 + lm) * ld + k0 + lh * 8;
  v8h lo = *(const v8h*)(base);
  v8h hi = *(const v8h*)(base + 16);
  return cat16(lo, hi);
}

// Stage an fp32 tile from global into padded f16 LDS.
template<int COLS, int STRIDE>
static __device__ inline void stage_tile(const float* __restrict__ g,
                                         _Float16* s, int nElem, int tid) {
  const float4* g4 = (const float4*)g;
  int n4 = nElem >> 2;
  for (int i = tid; i < n4; i += 256) {
    float4 v = g4[i];
    int e = i << 2;
    int r = e / COLS;
    int c = e - r * COLS;
    v4h h = { (_Float16)v.x, (_Float16)v.y, (_Float16)v.z, (_Float16)v.w };
    *(v4h*)(s + (size_t)r * STRIDE + c) = h;
  }
}

// Tiled GEMM: relu(A(64xK) * B(KxNout) + bias) -> f16 LDS (opt. transposed).
// A in LDS (padded), B^T f16 in global (L2-resident weights).
template<int KDIM, int NTILES, bool TRANS>
static __device__ inline void gemm_relu(const _Float16* sA, int ldA,
                                        const _Float16* __restrict__ gBT,
                                        const float* __restrict__ bias,
                                        _Float16* sD, int ldD, int tid) {
  int wave = tid >> 5, lane = tid & 31, lh = lane >> 4, lm = lane & 15;
  for (int t = wave; t < 4 * NTILES; t += 8) {
    int mt = t & 3, nt = t >> 2;
    v8f acc = V8F_ZERO;
#pragma unroll
    for (int k0 = 0; k0 < KDIM; k0 += 32) {
      v16h a = load_afrag(sA, ldA, mt * 16, k0, lh, lm);
      v16h b = load_bfrag(gBT, KDIM, nt * 16, k0, lh, lm);
      acc = wmma_f16(a, b, acc);
    }
    int col = nt * 16 + lm;
    float bc = bias[col];
#pragma unroll
    for (int r = 0; r < 8; ++r) {
      int row = mt * 16 + r + 8 * lh;                 // C layout: M=r+8*(lane/16)
      float v = fmaxf(acc[r] + bc, 0.f);
      if (TRANS) sD[(size_t)col * ldD + row] = (_Float16)v;
      else       sD[(size_t)row * ldD + col] = (_Float16)v;
    }
  }
}

// ---------------------------------------------------------------------------
// Weight prep: f32 -> f16 transpose into workspace (B^T layouts).
// ws halfs: WqT[0,65536) WkT[65536,131072) WvT[131072,147456) WoT[147456,163840)
// ---------------------------------------------------------------------------
__global__ void prep_weights(const float* __restrict__ Wq, const float* __restrict__ Wk,
                             const float* __restrict__ Wv, const float* __restrict__ Wo,
                             _Float16* __restrict__ wT) {
  int i = blockIdx.x * 256 + threadIdx.x;
  if (i < 65536) {
    int n = i >> 8, k = i & 255;
    wT[i] = (_Float16)Wq[k * 256 + n];
  } else if (i < 131072) {
    int j = i - 65536; int n = j >> 8, k = j & 255;
    wT[i] = (_Float16)Wk[k * 256 + n];
  } else if (i < 147456) {
    int j = i - 131072; int n = j >> 7, k = j & 127;
    wT[i] = (_Float16)Wv[k * 128 + n];
  } else if (i < 163840) {
    int j = i - 147456; int n = j >> 7, k = j & 127;
    wT[i] = (_Float16)Wo[k * 128 + n];
  }
}

// ---------------------------------------------------------------------------
// Main kernel: one block per (b,n); 8 waves; wave == head in attention phase.
// LDS (halfs): sQ[0,16896) sK[16896,33792) sVT[33792,43008)
//              sU[43008,60928)  (staging | {sAw strips 9216, sO 8704})
//              sMask bytes at half-offset 60928 (4096 B).  Total 125952 B.
// ---------------------------------------------------------------------------
#define SMEM_BYTES 125952

__global__ __launch_bounds__(256) void
attn_kernel(const float* __restrict__ q_in, const float* __restrict__ k_in,
            const float* __restrict__ v_in, const unsigned char* __restrict__ mask,
            const _Float16* __restrict__ wT,
            const float* __restrict__ bq, const float* __restrict__ bk,
            const float* __restrict__ bv, const float* __restrict__ bo,
            float* __restrict__ out, float* __restrict__ attn_out) {
  extern __shared__ __align__(16) char smem[];
  _Float16* sQ  = (_Float16*)smem;          // 64 x 256, ld 264
  _Float16* sK  = sQ + 16896;               // 64 x 256, ld 264
  _Float16* sVT = sK + 16896;               // 128 x 64 (v^T), ld 72
  _Float16* sU  = sVT + 9216;               // staging / attn strips
  _Float16* sO  = sU + 9216;                // 64 x 128, ld 136
  unsigned char* sMask = (unsigned char*)(sU + 17920);  // 64x64 bytes

  const int tid = threadIdx.x, bn = blockIdx.x;
  const int wave = tid >> 5, lane = tid & 31, lh = lane >> 4, lm = lane & 15;

  const _Float16* WqT = wT;
  const _Float16* WkT = wT + 65536;
  const _Float16* WvT = wT + 131072;
  const _Float16* WoT = wT + 147456;

  // ---- Q projection ----
  stage_tile<256, 264>(q_in + (size_t)bn * 16384, sU, 16384, tid);
  __syncthreads();
  gemm_relu<256, 16, false>(sU, 264, WqT, bq, sQ, 264, tid);
  __syncthreads();

  // ---- K projection (+ stage mask tile) ----
  stage_tile<256, 264>(k_in + (size_t)bn * 16384, sU, 16384, tid);
  ((uint4*)sMask)[tid] = ((const uint4*)(mask + (size_t)bn * 4096))[tid];
  __syncthreads();
  gemm_relu<256, 16, false>(sU, 264, WkT, bk, sK, 264, tid);
  __syncthreads();

  // ---- V projection (stored transposed: sVT[d][s]) ----
  stage_tile<128, 136>(v_in + (size_t)bn * 8192, sU, 8192, tid);
  __syncthreads();
  gemm_relu<128, 8, true>(sU, 136, WvT, bv, sVT, 72, tid);
  __syncthreads();

  // ---- Attention: one head per wave (H == 8 waves) ----
  {
    const int h = wave;
    _Float16* strip = sU + wave * (16 * 72);       // per-wave 16x64 attn strip
    const float scale = 0.17677669529663687f;      // 1/sqrt(32)
    const float NEG_INF = -__builtin_inff();
    float* attnb = attn_out + (((size_t)bn * 8 + h) * 64) * 64;

    for (int mt = 0; mt < 4; ++mt) {
      // scores: q_h(16x32) x k_h^T(32x64) -> 4 tiles, one WMMA each (K=dk=32)
      v16h aq = load_afrag(sQ, 264, mt * 16, h * 32, lh, lm);
      float vals[4][8];
#pragma unroll
      for (int st = 0; st < 4; ++st) {
        v16h bkf = load_bfrag(sK, 264, st * 16, h * 32, lh, lm);
        v8f acc = wmma_f16(aq, bkf, V8F_ZERO);
#pragma unroll
        for (int r = 0; r < 8; ++r) {
          int tr = mt * 16 + r + 8 * lh;
          int sc = st * 16 + lm;
          float x = acc[r] * scale;
          if (sMask[tr * 64 + sc]) x = NEG_INF;
          vals[st][r] = x;
        }
      }
      // softmax over 64 cols: per-lane 4 partials + xor-reduce over 16 lanes
#pragma unroll
      for (int r = 0; r < 8; ++r) {
        float rmax = fmaxf(fmaxf(vals[0][r], vals[1][r]),
                           fmaxf(vals[2][r], vals[3][r]));
#pragma unroll
        for (int off = 8; off >= 1; off >>= 1)
          rmax = fmaxf(rmax, __shfl_xor(rmax, off, 32));
        float rsum = 0.f;
#pragma unroll
        for (int st = 0; st < 4; ++st) {
          float p = __expf(vals[st][r] - rmax);
          vals[st][r] = p;
          rsum += p;
        }
#pragma unroll
        for (int off = 8; off >= 1; off >>= 1)
          rsum += __shfl_xor(rsum, off, 32);
        float inv = 1.0f / rsum;
#pragma unroll
        for (int st = 0; st < 4; ++st) vals[st][r] *= inv;
      }
      // emit attn (f32 global) + strip (f16 LDS, A-matrix for attn @ v)
#pragma unroll
      for (int st = 0; st < 4; ++st)
#pragma unroll
        for (int r = 0; r < 8; ++r) {
          int tr = mt * 16 + r + 8 * lh;
          int sc = st * 16 + lm;
          float p = vals[st][r];
          attnb[(size_t)tr * 64 + sc] = p;
          strip[(r + 8 * lh) * 72 + sc] = (_Float16)p;
        }
      // out tile: attn(16x64) @ v_h(64x16), K=64 -> 2 WMMAs
      v8f oacc = V8F_ZERO;
#pragma unroll
      for (int kt = 0; kt < 2; ++kt) {
        v16h aa = load_afrag(strip, 72, 0, kt * 32, lh, lm);
        v16h bb = load_bfrag(sVT, 72, h * 16, kt * 32, lh, lm);
        oacc = wmma_f16(aa, bb, oacc);
      }
#pragma unroll
      for (int r = 0; r < 8; ++r) {
        int tr = mt * 16 + r + 8 * lh;
        sO[(size_t)tr * 136 + h * 16 + lm] = (_Float16)oacc[r];
      }
    }
  }
  __syncthreads();

  // ---- Output projection: relu(O(64x128) @ Wo + bo) -> global f32 ----
  for (int t = wave; t < 32; t += 8) {
    int mt = t & 3, nt = t >> 2;
    v8f acc = V8F_ZERO;
#pragma unroll
    for (int k0 = 0; k0 < 128; k0 += 32) {
      v16h a = load_afrag(sO, 136, mt * 16, k0, lh, lm);
      v16h b = load_bfrag(WoT, 128, nt * 16, k0, lh, lm);
      acc = wmma_f16(a, b, acc);
    }
    int col = nt * 16 + lm;
    float bc = bo[col];
#pragma unroll
    for (int r = 0; r < 8; ++r) {
      int row = mt * 16 + r + 8 * lh;
      out[(size_t)bn * 8192 + row * 128 + col] = fmaxf(acc[r] + bc, 0.f);
    }
  }
}

// ---------------------------------------------------------------------------
extern "C" void kernel_launch(void* const* d_in, const int* in_sizes, int n_in,
                              void* d_out, int out_size, void* d_ws, size_t ws_size,
                              hipStream_t stream) {
  (void)in_sizes; (void)n_in; (void)out_size; (void)ws_size;
  const float* query = (const float*)d_in[0];
  const float* key   = (const float*)d_in[1];
  const float* value = (const float*)d_in[2];
  const unsigned char* mask = (const unsigned char*)d_in[3];
  const float* Wq = (const float*)d_in[4];
  const float* bq = (const float*)d_in[5];
  const float* Wk = (const float*)d_in[6];
  const float* bk = (const float*)d_in[7];
  const float* Wv = (const float*)d_in[8];
  const float* bv = (const float*)d_in[9];
  const float* Wo = (const float*)d_in[10];
  const float* bo = (const float*)d_in[11];

  _Float16* wT = (_Float16*)d_ws;          // 163840 halfs = 320 KB
  float* out = (float*)d_out;              // 16*128*64*128 = 16777216 floats
  float* attn = out + 16777216;            // 16*128*8*64*64 = 67108864 floats

  prep_weights<<<640, 256, 0, stream>>>(Wq, Wk, Wv, Wo, wT);

  (void)hipFuncSetAttribute(reinterpret_cast<const void*>(attn_kernel),
                            hipFuncAttributeMaxDynamicSharedMemorySize, SMEM_BYTES);
  attn_kernel<<<2048, 256, SMEM_BYTES, stream>>>(query, key, value, mask, wT,
                                                 bq, bk, bv, bo, out, attn);
}
